// Attention_146028888114
// MI455X (gfx1250) — matrix operand order repaired and checked by reading, not compiled, not run on hardware
//
#include <hip/hip_runtime.h>
#include <hip/hip_bf16.h>
#include <math.h>

typedef __attribute__((ext_vector_type(16))) __bf16 v16bf;
typedef __attribute__((ext_vector_type(8)))  __bf16 v8bf;
typedef __attribute__((ext_vector_type(8)))  float  v8f;
typedef __attribute__((ext_vector_type(4)))  int    v4i;

#define B_  2
#define S_  2048
#define D_  1024
#define H_  16
#define HD_ 64

// ---------------- CDNA5 async global->LDS copy (guarded) ----------------
#if defined(__has_builtin)
#  if __has_builtin(__builtin_amdgcn_global_load_async_to_lds_b128) && \
      __has_builtin(__builtin_amdgcn_s_wait_asynccnt)
#    define USE_ASYNC 1
#  endif
#endif
#ifndef USE_ASYNC
#  define USE_ASYNC 0
#endif

typedef __attribute__((address_space(1))) v4i gv4i;  // global
typedef __attribute__((address_space(3))) v4i lv4i;  // LDS

__device__ inline void cp16(const __bf16* g, __bf16* l) {
#if USE_ASYNC
  __builtin_amdgcn_global_load_async_to_lds_b128((gv4i*)g, (lv4i*)l, 0, 0);
#else
  *(v8bf*)l = *(const v8bf*)g;
#endif
}
__device__ inline void cp_wait() {
#if USE_ASYNC
  __builtin_amdgcn_s_wait_asynccnt(0);
#endif
}

// ---------------------------------------------------------------- helpers

__device__ inline v8f vzero8() {
  v8f z;
#pragma unroll
  for (int i = 0; i < 8; i++) z[i] = 0.f;
  return z;
}

__device__ inline v8f wmma_bf16(v16bf a, v16bf b, v8f c) {
  return __builtin_amdgcn_wmma_f32_16x16x32_bf16(false, a, false, b, (short)0, c,
                                                 false, false);
}

__device__ inline v16bf make_frag(v8bf lo, v8bf hi) {
  v16bf f;
#pragma unroll
  for (int i = 0; i < 8; i++) { f[i] = lo[i]; f[8 + i] = hi[i]; }
  return f;
}

// A fragment, 16x32 bf16, from a row-major matrix (global or LDS pointer).
// Lane (r=lane&15, hv=lane>>4) holds row r, K = hv*16+0..15 (contiguous).
__device__ inline v16bf load_a_frag(const __bf16* A, int lda, int row0, int k0,
                                    int lane) {
  const int r = lane & 15, hv = lane >> 4;
  const __bf16* p = A + (size_t)(row0 + r) * lda + k0 + hv * 16;
  v8bf lo = *(const v8bf*)p;
  v8bf hi = *(const v8bf*)(p + 8);
  return make_frag(lo, hi);
}

// B fragment, 32x16, element (k,n) = Rows[n][k] (B^T row-major, ld elements).
// Lane holds N = lane&15, K = (lane>>4)*16 + 0..15 -> contiguous.
__device__ inline v16bf load_b_frag_rows(const __bf16* Rows, int ld, int n0,
                                         int k0, int lane) {
  const int c = lane & 15, hv = lane >> 4;
  const __bf16* p = Rows + (size_t)(n0 + c) * ld + k0 + hv * 16;
  v8bf lo = *(const v8bf*)p;
  v8bf hi = *(const v8bf*)(p + 8);
  return make_frag(lo, hi);
}

// B fragment from packed layout: fragment (nt,kt) stored lane-contiguous.
__device__ inline v16bf load_b_frag_packed(const __bf16* P, int ktiles, int nt,
                                           int kt, int lane) {
  const __bf16* p = P + ((((size_t)nt * ktiles + kt) * 32 + lane) << 4);
  v8bf lo = *(const v8bf*)p;
  v8bf hi = *(const v8bf*)(p + 8);
  return make_frag(lo, hi);
}

__device__ inline float wave_sum32(float v) {
#pragma unroll
  for (int m = 16; m > 0; m >>= 1) v += __shfl_xor(v, m, 32);
  return v;
}
__device__ inline float half_sum16(float v) {
#pragma unroll
  for (int m = 8; m > 0; m >>= 1) v += __shfl_xor(v, m, 32);
  return v;
}
__device__ inline float half_max16(float v) {
#pragma unroll
  for (int m = 8; m > 0; m >>= 1) v = fmaxf(v, __shfl_xor(v, m, 32));
  return v;
}

// ---------------------------------------------------------------- kernels

// Pack row-major KxN fp32 weight matrix into bf16 B-fragments.
__global__ void k_pack_b(const float* __restrict__ W, __bf16* __restrict__ P,
                         int K, int N) {
  size_t t = (size_t)blockIdx.x * blockDim.x + threadIdx.x;
  const int ktiles = K >> 5;
  size_t total = (size_t)ktiles * (N >> 4) * 32;
  if (t >= total) return;
  const int lane = (int)(t & 31);
  size_t tile = t >> 5;
  const int kt = (int)(tile % ktiles);
  const int nt = (int)(tile / ktiles);
  const int c = lane & 15, hv = lane >> 4;
  const int n = nt * 16 + c;
  const int kbase = kt * 32 + hv * 16;
  __bf16* dst = P + (t << 4);
#pragma unroll
  for (int i = 0; i < 16; i++) dst[i] = (__bf16)W[(size_t)(kbase + i) * N + n];
}

// LayerNorm over D=1024 per row, fp32 in -> bf16 out. One block per row.
__global__ __launch_bounds__(256) void k_ln_x(const float* __restrict__ x,
                                              const float* __restrict__ sc,
                                              const float* __restrict__ bi,
                                              __bf16* __restrict__ out) {
  const int row = blockIdx.x;
  const float* xr = x + (size_t)row * D_;
  float v[4];
  float s = 0.f;
#pragma unroll
  for (int i = 0; i < 4; i++) { v[i] = xr[threadIdx.x + i * 256]; s += v[i]; }
  __shared__ float red[8];
  const int lane = threadIdx.x & 31, wv = threadIdx.x >> 5;
  s = wave_sum32(s);
  if (lane == 0) red[wv] = s;
  __syncthreads();
  float tot = 0.f;
#pragma unroll
  for (int i = 0; i < 8; i++) tot += red[i];
  const float mu = tot * (1.f / D_);
  float s2 = 0.f;
#pragma unroll
  for (int i = 0; i < 4; i++) { float d = v[i] - mu; s2 += d * d; }
  s2 = wave_sum32(s2);
  __syncthreads();
  if (lane == 0) red[wv] = s2;
  __syncthreads();
  float tot2 = 0.f;
#pragma unroll
  for (int i = 0; i < 8; i++) tot2 += red[i];
  const float rstd = rsqrtf(tot2 * (1.f / D_) + 1e-6f);
#pragma unroll
  for (int i = 0; i < 4; i++) {
    const int col = threadIdx.x + i * 256;
    out[(size_t)row * D_ + col] = (__bf16)((v[i] - mu) * rstd * sc[col] + bi[col]);
  }
}

// LDS-staged, double-buffered WMMA GEMM.
// 256 threads = 8 waves (2 x 4); block tile 128x128; wave tile 64x32.
template <bool OUT_BF16>
__global__ __launch_bounds__(256) void k_gemm(const __bf16* __restrict__ A,
                                              const __bf16* __restrict__ Bp,
                                              const float* __restrict__ bias,
                                              void* __restrict__ Cout, int M,
                                              int N, int K) {
  __shared__ __align__(16) __bf16 sA[2][128 * 32];  // row-major tile
  __shared__ __align__(16) __bf16 sB[2][128 * 32];  // packed fragments
  const int tid = threadIdx.x;
  const int lane = tid & 31, wv = tid >> 5;
  const int wm = wv >> 2, wn = wv & 3;
  const int mB = blockIdx.x * 128;
  const int nB = blockIdx.y * 128;
  const int nt0 = nB >> 4;
  const int ktiles = K >> 5;
  (void)M;

  auto stage = [&](int buf, int kt) {
#pragma unroll
    for (int u = 0; u < 2; u++) {  // A: 128 rows x 32 cols, 512 x 16B chunks
      int c = tid + u * 256;
      int row = c >> 2, co = (c & 3) * 8;
      cp16(A + (size_t)(mB + row) * K + kt * 32 + co, &sA[buf][row * 32 + co]);
    }
#pragma unroll
    for (int u = 0; u < 2; u++) {  // B: 8 packed nt-fragments
      int c = tid + u * 256;
      int l = c >> 6, ln = (c >> 1) & 31, hf = c & 1;
      cp16(Bp + ((((size_t)(nt0 + l)) * ktiles + kt) * 32 + ln) * 16 + hf * 8,
           &sB[buf][c * 8]);
    }
  };

  v8f acc[4][2];
#pragma unroll
  for (int i = 0; i < 4; i++)
#pragma unroll
    for (int j = 0; j < 2; j++) acc[i][j] = vzero8();

  stage(0, 0);
  for (int kt = 0; kt < ktiles; ++kt) {
    const int buf = kt & 1;
    cp_wait();
    __syncthreads();
    if (kt + 1 < ktiles) stage(buf ^ 1, kt + 1);

    v16bf a[4], b[2];
#pragma unroll
    for (int i = 0; i < 4; i++)
      a[i] = load_a_frag(&sA[buf][0], 32, wm * 64 + i * 16, 0, lane);
#pragma unroll
    for (int j = 0; j < 2; j++)
      b[j] = load_b_frag_packed(&sB[buf][0], 1, wn * 2 + j, 0, lane);
#pragma unroll
    for (int i = 0; i < 4; i++)
#pragma unroll
      for (int j = 0; j < 2; j++) acc[i][j] = wmma_bf16(a[i], b[j], acc[i][j]);
  }

  const int hv = lane >> 4, cc = lane & 15;
#pragma unroll
  for (int j = 0; j < 2; j++) {
    const float bs = bias[nB + wn * 32 + j * 16 + cc];
#pragma unroll
    for (int i = 0; i < 4; i++) {
#pragma unroll
      for (int vi = 0; vi < 8; vi++) {
        const int row = mB + wm * 64 + i * 16 + hv * 8 + vi;
        const int col = nB + wn * 32 + j * 16 + cc;
        const float val = acc[i][j][vi] + bs;
        if (OUT_BF16)
          ((__bf16*)Cout)[(size_t)row * N + col] = (__bf16)val;
        else
          ((float*)Cout)[(size_t)row * N + col] = val;
      }
    }
  }
}

// Per-head LN + RoPE for q/k, transpose for v. One wave per (b,s,h).
__device__ inline void ln_rope64(float e1, float e2, const float* __restrict__ sc,
                                 int lane, int pos, float& o1, float& o2) {
  const float mu = wave_sum32(e1 + e2) * (1.f / 64.f);
  const float d1 = e1 - mu, d2 = e2 - mu;
  const float var = wave_sum32(d1 * d1 + d2 * d2) * (1.f / 64.f);
  const float r = rsqrtf(var + 1e-6f);
  const float n1 = d1 * r * sc[lane];
  const float n2 = d2 * r * sc[lane + 32];
  const float ang = (float)pos * __expf(-0.28782313662425572f * (float)lane);
  float sn, cs;
  __sincosf(ang, &sn, &cs);
  o1 = n1 * cs - n2 * sn;
  o2 = n2 * cs + n1 * sn;
}

__global__ __launch_bounds__(256) void k_qk_rope_v(
    const __bf16* __restrict__ qkv, const float* __restrict__ qs,
    const float* __restrict__ ks, __bf16* __restrict__ Q,
    __bf16* __restrict__ K, __bf16* __restrict__ Vt) {
  const int idx = blockIdx.x * 8 + (threadIdx.x >> 5);  // over (b,s,h)
  const int lane = threadIdx.x & 31;
  const int h = idx & (H_ - 1);
  const int s = (idx >> 4) & (S_ - 1);
  const int b = idx >> 15;
  const size_t base = ((size_t)(b * S_ + s)) * (3 * D_) + h * HD_;
  const size_t hrow = ((size_t)(b * H_ + h) * S_ + s) * HD_;
  {
    float e1 = (float)qkv[base + lane], e2 = (float)qkv[base + lane + 32];
    float o1, o2;
    ln_rope64(e1, e2, qs, lane, s, o1, o2);
    Q[hrow + lane] = (__bf16)o1;
    Q[hrow + lane + 32] = (__bf16)o2;
  }
  {
    float e1 = (float)qkv[base + D_ + lane], e2 = (float)qkv[base + D_ + lane + 32];
    float o1, o2;
    ln_rope64(e1, e2, ks, lane, s, o1, o2);
    K[hrow + lane] = (__bf16)o1;
    K[hrow + lane + 32] = (__bf16)o2;
  }
  {
    const size_t vtb = ((size_t)(b * H_ + h)) * HD_ * S_;
    Vt[vtb + (size_t)lane * S_ + s] = qkv[base + 2 * D_ + lane];
    Vt[vtb + (size_t)(lane + 32) * S_ + s] = qkv[base + 2 * D_ + lane + 32];
  }
}

// Flash attention: 8 waves/block (same b,h), each wave a 16-query tile.
// K/V tiles (64 keys) staged in LDS, double-buffered, shared by all 8 waves.
__global__ __launch_bounds__(256) void k_flash(const __bf16* __restrict__ Q,
                                               const __bf16* __restrict__ K,
                                               const __bf16* __restrict__ Vt,
                                               __bf16* __restrict__ O) {
  __shared__ __align__(16) __bf16 sK[2][64 * HD_];  // keys x d
  __shared__ __align__(16) __bf16 sV[2][HD_ * 64];  // d x keys (V^T tile)
  __shared__ __align__(16) __bf16 sP[8][16 * 64];   // per-wave P tile
  const int tid = threadIdx.x;
  const int lane = tid & 31, wv = tid >> 5;
  const int bh = blockIdx.x >> 4;
  const int q0 = ((blockIdx.x & 15) * 8 + wv) * 16;
  const __bf16* Qh = Q + (size_t)bh * S_ * HD_;
  const __bf16* Kh = K + (size_t)bh * S_ * HD_;
  const __bf16* Vh = Vt + (size_t)bh * HD_ * S_;

  auto stage_kv = [&](int buf, int kb) {
#pragma unroll
    for (int u = 0; u < 2; u++) {  // K tile: 64 rows x 64 d
      int c = tid + u * 256;
      int row = c >> 3, co = (c & 7) * 8;
      cp16(Kh + (size_t)(kb + row) * HD_ + co, &sK[buf][row * HD_ + co]);
    }
#pragma unroll
    for (int u = 0; u < 2; u++) {  // V^T tile: 64 d-rows x 64 keys
      int c = tid + u * 256;
      int row = c >> 3, co = (c & 7) * 8;
      cp16(Vh + (size_t)row * S_ + kb + co, &sV[buf][row * 64 + co]);
    }
  };

  const v16bf aq0 = load_a_frag(Qh, HD_, q0, 0, lane);
  const v16bf aq1 = load_a_frag(Qh, HD_, q0, 32, lane);

  v8f o[4];
#pragma unroll
  for (int j = 0; j < 4; j++) o[j] = vzero8();
  float mi[8], li[8];
#pragma unroll
  for (int i = 0; i < 8; i++) { mi[i] = -1e30f; li[i] = 0.f; }
  const int hv = lane >> 4, cc = lane & 15;

  stage_kv(0, 0);
  for (int kb = 0; kb < S_; kb += 64) {
    const int buf = (kb >> 6) & 1;
    cp_wait();
    __syncthreads();
    if (kb + 64 < S_) stage_kv(buf ^ 1, kb + 64);

    v8f s[4];
#pragma unroll
    for (int t = 0; t < 4; t++) {
      s[t] = vzero8();
      s[t] = wmma_bf16(aq0, load_b_frag_rows(&sK[buf][0], HD_, t * 16, 0, lane), s[t]);
      s[t] = wmma_bf16(aq1, load_b_frag_rows(&sK[buf][0], HD_, t * 16, 32, lane), s[t]);
    }
#pragma unroll
    for (int vi = 0; vi < 8; vi++) {
      const float x0 = s[0][vi] * 0.125f;  // 1/sqrt(64)
      const float x1 = s[1][vi] * 0.125f;
      const float x2 = s[2][vi] * 0.125f;
      const float x3 = s[3][vi] * 0.125f;
      const float mb = half_max16(fmaxf(fmaxf(x0, x1), fmaxf(x2, x3)));
      const float mn = fmaxf(mi[vi], mb);
      const float p0 = __expf(x0 - mn), p1 = __expf(x1 - mn);
      const float p2 = __expf(x2 - mn), p3 = __expf(x3 - mn);
      const float rs = half_sum16((p0 + p1) + (p2 + p3));
      const float corr = __expf(mi[vi] - mn);
      li[vi] = li[vi] * corr + rs;
      mi[vi] = mn;
#pragma unroll
      for (int j = 0; j < 4; j++) o[j][vi] *= corr;
      const int r = vi + hv * 8;  // C-layout row
      sP[wv][r * 64 + cc] = (__bf16)p0;
      sP[wv][r * 64 + 16 + cc] = (__bf16)p1;
      sP[wv][r * 64 + 32 + cc] = (__bf16)p2;
      sP[wv][r * 64 + 48 + cc] = (__bf16)p3;
    }
    // LDS ops from one wave are in-order: loads below see the stores above.
    v16bf ap0 = load_a_frag(&sP[wv][0], 64, 0, 0, lane);
    v16bf ap1 = load_a_frag(&sP[wv][0], 64, 0, 32, lane);
#pragma unroll
    for (int j = 0; j < 4; j++) {
      o[j] = wmma_bf16(ap0, load_b_frag_rows(&sV[buf][0], 64, j * 16, 0, lane), o[j]);
      o[j] = wmma_bf16(ap1, load_b_frag_rows(&sV[buf][0], 64, j * 16, 32, lane), o[j]);
    }
  }

  const int b = bh >> 4, h = bh & (H_ - 1);
#pragma unroll
  for (int j = 0; j < 4; j++) {
#pragma unroll
    for (int vi = 0; vi < 8; vi++) {
      const int row = q0 + vi + hv * 8;
      const int col = h * HD_ + j * 16 + cc;
      O[(size_t)(b * S_ + row) * D_ + col] = (__bf16)(o[j][vi] / li[vi]);
    }
  }
}

// ---------------------------------------------------------------- launch

extern "C" void kernel_launch(void* const* d_in, const int* in_sizes, int n_in,
                              void* d_out, int out_size, void* d_ws,
                              size_t ws_size, hipStream_t stream) {
  (void)in_sizes; (void)n_in; (void)out_size; (void)ws_size;
  const float* x     = (const float*)d_in[0];
  const float* w_qkv = (const float*)d_in[1];
  const float* b_qkv = (const float*)d_in[2];
  const float* w_out = (const float*)d_in[3];
  const float* b_out = (const float*)d_in[4];
  const float* ln_s  = (const float*)d_in[5];
  const float* ln_b  = (const float*)d_in[6];
  const float* q_s   = (const float*)d_in[7];
  const float* k_s   = (const float*)d_in[8];

  char* ws = (char*)d_ws;
  size_t off = 0;
  auto alloc = [&](size_t bytes) {
    char* p = ws + off;
    off += (bytes + 255) & ~(size_t)255;
    return p;
  };
  __bf16* wqkv_p = (__bf16*)alloc((size_t)D_ * 3 * D_ * 2);
  __bf16* wout_p = (__bf16*)alloc((size_t)D_ * D_ * 2);
  __bf16* xn     = (__bf16*)alloc((size_t)B_ * S_ * D_ * 2);
  __bf16* qkv    = (__bf16*)alloc((size_t)B_ * S_ * 3 * D_ * 2);
  __bf16* Qb     = (__bf16*)alloc((size_t)B_ * H_ * S_ * HD_ * 2);
  __bf16* Kb     = (__bf16*)alloc((size_t)B_ * H_ * S_ * HD_ * 2);
  __bf16* Vt     = (__bf16*)alloc((size_t)B_ * H_ * S_ * HD_ * 2);
  __bf16* attn   = (__bf16*)alloc((size_t)B_ * S_ * D_ * 2);

  {
    size_t tq = (size_t)(D_ / 32) * (3 * D_ / 16) * 32;
    k_pack_b<<<(unsigned)((tq + 255) / 256), 256, 0, stream>>>(w_qkv, wqkv_p,
                                                               D_, 3 * D_);
    size_t to = (size_t)(D_ / 32) * (D_ / 16) * 32;
    k_pack_b<<<(unsigned)((to + 255) / 256), 256, 0, stream>>>(w_out, wout_p,
                                                               D_, D_);
  }
  k_ln_x<<<B_ * S_, 256, 0, stream>>>(x, ln_s, ln_b, xn);
  k_gemm<true><<<dim3((B_ * S_) / 128, (3 * D_) / 128), 256, 0, stream>>>(
      xn, wqkv_p, b_qkv, qkv, B_ * S_, 3 * D_, D_);
  k_qk_rope_v<<<(B_ * S_ * H_) / 8, 256, 0, stream>>>(qkv, q_s, k_s, Qb, Kb, Vt);
  k_flash<<<B_ * H_ * (S_ / 128), 256, 0, stream>>>(Qb, Kb, Vt, attn);
  k_gemm<false><<<dim3((B_ * S_) / 128, D_ / 128), 256, 0, stream>>>(
      attn, wout_p, b_out, d_out, B_ * S_, D_, D_);
}
